// ClusteredAttention_90761248899356
// MI455X (gfx1250) — matrix operand order repaired and checked by reading, not compiled, not run on hardware
//
#include <hip/hip_runtime.h>
#include <math.h>

// ---------------------------------------------------------------------------
// Types for CDNA5 WMMA (wave32, 16x16x32 bf16 -> f32)
// ---------------------------------------------------------------------------
typedef __bf16 bf16t;
typedef __attribute__((ext_vector_type(16))) __bf16 v16bf;
typedef __attribute__((ext_vector_type(8)))  __bf16 v8bf;
typedef __attribute__((ext_vector_type(8)))  float  v8f;
typedef __attribute__((ext_vector_type(4)))  int    v4i;
typedef __attribute__((address_space(1))) v4i gv4i;   // global int4
typedef __attribute__((address_space(3))) v4i lv4i;   // LDS int4

#define DEVINL __device__ __forceinline__

// Problem constants (reference: B=8, C=4096, E=512, H=4, D=128, NC=64, WSZ=128)
static constexpr int B_   = 8;
static constexpr int C_   = 4096;
static constexpr int E_   = 512;
static constexpr int H_   = 4;
static constexpr int D_   = 128;
static constexpr int NC_  = 64;
static constexpr int WSZ_ = 128;
static constexpr int HD_  = H_ * D_;    // 512
static constexpr int TWOC = 2 * C_;     // 8192
static constexpr int MTOK = B_ * C_;    // 32768

DEVINL v8f wmma_bf16(v16bf a, v16bf b, v8f c) {
  // (neg_a, A, neg_b, B, c_mod, C, reuse_a, reuse_b)
  return __builtin_amdgcn_wmma_f32_16x16x32_bf16(false, a, false, b, (short)0, c, false, false);
}

// A-fragment (16x32 bf16): lane holds row m = lane&15; K-half kh = lane>>4.
// Elements 0..15 <- K = kh*16+0..15
DEVINL v16bf a_frag(const bf16t* rowp, int lane) {
  int kh = (lane >> 4) & 1;
  const bf16t* p = rowp + kh * 16;
  v8bf lo = *(const v8bf*)(p);
  v8bf hi = *(const v8bf*)(p + 8);
  return __builtin_shufflevector(lo, hi, 0,1,2,3,4,5,6,7,8,9,10,11,12,13,14,15);
}

// B-fragment (32x16 bf16): lane holds column n = lane&15; K run = (lane>>4)*16 + 0..15.
// colp: this lane's column (N-major / K-contiguous storage), already offset by k0.
DEVINL v16bf b_frag(const bf16t* colp, int lane) {
  int kh = (lane >> 4) & 1;
  return *(const v16bf*)(colp + kh * 16);
}

// ---------------------------------------------------------------------------
// Async global->LDS copy (CDNA5 GLOBAL_LOAD_ASYNC_TO_LDS_B128, ASYNCcnt
// tracked). Guarded so the file still compiles if the builtin is unavailable.
// ---------------------------------------------------------------------------
#if __has_builtin(__builtin_amdgcn_global_load_async_to_lds_b128)
#define HAVE_ASYNC_LDS 1
#else
#define HAVE_ASYNC_LDS 0
#endif

DEVINL void async_copy16(const bf16t* gsrc, bf16t* ldst) {
#if HAVE_ASYNC_LDS
  __builtin_amdgcn_global_load_async_to_lds_b128((gv4i*)gsrc, (lv4i*)ldst, 0, 0);
#else
  *(v8bf*)ldst = *(const v8bf*)gsrc;
#endif
}

DEVINL void async_wait_all() {
#if HAVE_ASYNC_LDS
#if __has_builtin(__builtin_amdgcn_s_wait_asynccnt)
  __builtin_amdgcn_s_wait_asynccnt(0);
#else
  asm volatile("s_wait_asynccnt 0" ::: "memory");
#endif
#endif
}

// ---------------------------------------------------------------------------
// Shared GEMM mainloop: one wave computes a 16x128 strip; the block's common
// 128-col B tile is streamed through a 2 x (128x128) LDS double buffer with
// async copies. K must be a multiple of 128; A/B leading dim == K.
// ---------------------------------------------------------------------------
DEVINL void gemm_mainloop_ldsB(const bf16t* __restrict__ arp,   // lane's A row
                               const bf16t* __restrict__ btile, // Bt + col0*K
                               int K, int tid, int lane, v8f acc[8]) {
  __shared__ bf16t lbuf[2][128 * 128];   // 64 KB double buffer
  const int ln = lane & 15;
  const int col = tid >> 1, seg = (tid & 1) * 64;  // 256 thr: 128 cols x 2 segs
  const int nchunks = K >> 7;

  {
    const bf16t* g = btile + (size_t)col * K + seg;
    bf16t* l = &lbuf[0][col * 128 + seg];
#pragma unroll
    for (int j = 0; j < 8; ++j) async_copy16(g + j * 8, l + j * 8);
  }
  async_wait_all();
  __syncthreads();

  for (int kc = 0; kc < nchunks; ++kc) {
    const int cur = kc & 1;
    if (kc + 1 < nchunks) {
      const bf16t* g = btile + (size_t)col * K + (kc + 1) * 128 + seg;
      bf16t* l = &lbuf[1 - cur][col * 128 + seg];
#pragma unroll
      for (int j = 0; j < 8; ++j) async_copy16(g + j * 8, l + j * 8);
      __builtin_prefetch(arp + (kc + 1) * 128, 0, 0);  // next A chunk
    }
    // all 4 A fragments of this chunk up-front (distinct regs -> pipelining)
    v16bf a[4];
#pragma unroll
    for (int u = 0; u < 4; ++u) a[u] = a_frag(arp + kc * 128 + u * 32, lane);
    const bf16t* bb = &lbuf[cur][0];
#pragma unroll
    for (int u = 0; u < 4; ++u)
#pragma unroll
      for (int t = 0; t < 8; ++t)
        acc[t] = wmma_bf16(a[u], b_frag(bb + (t * 16 + ln) * 128 + u * 32, lane), acc[t]);
    if (kc + 1 < nchunks) async_wait_all();
    __syncthreads();
  }
}

// ---------------------------------------------------------------------------
// 0) zero numer / denom-count / aux accumulator
// ---------------------------------------------------------------------------
__global__ void init_zero_kernel(float* numer, float* cnt, float* aux) {
  size_t idx = (size_t)blockIdx.x * 256 + threadIdx.x;
  if (idx < (size_t)B_ * H_ * C_ * D_) numer[idx] = 0.f;
  if (idx < (size_t)B_ * H_ * C_)      cnt[idx]   = 0.f;
  if (idx == 0)                        aux[0]     = 0.f;
}

// ---------------------------------------------------------------------------
// 1) cast weights (transposed to N-major for B-fragments) + means to bf16
// ---------------------------------------------------------------------------
__global__ void cast_weights_kernel(const float* Wq, const float* Wk, const float* Wv,
                                    const float* Wu, const float* means,
                                    bf16t* wqt, bf16t* wkt, bf16t* wvt, bf16t* wut,
                                    bf16t* meansbf) {
  int idx = blockIdx.x * 256 + threadIdx.x;          // 4*262144 + 32768 total
  if (idx < 4 * 262144) {
    int m = idx >> 18;
    int r = idx & 262143;
    int n = r >> 9, k = r & 511;
    const float* W = (m == 0) ? Wq : (m == 1) ? Wk : (m == 2) ? Wv : Wu;
    bf16t* O = (m == 0) ? wqt : (m == 1) ? wkt : (m == 2) ? wvt : wut;
    O[((size_t)n << 9) + k] = (bf16t)W[((size_t)k << 9) + n];
  } else {
    int j = idx - 4 * 262144;                         // H*NC*D = 32768
    meansbf[j] = (bf16t)means[j];
  }
}

__global__ void cast_x_kernel(const float* x, bf16t* xbf) {
  size_t idx = (size_t)blockIdx.x * 256 + threadIdx.x;   // exactly MTOK*E
  xbf[idx] = (bf16t)x[idx];
}

// ---------------------------------------------------------------------------
// 2) QKV projection GEMM: [32768 x 512] x [512 x 512] per matrix.
//    8 waves/block; wave = 16 rows x 128 cols (= one full head of D).
//    Epilogue: bf16 q/k/v + in-register row norms -> normalized xn (bf16).
// ---------------------------------------------------------------------------
__global__ __launch_bounds__(256) void gemm_qkv_kernel(
    const bf16t* __restrict__ xbf,
    const bf16t* __restrict__ wqt, const bf16t* __restrict__ wkt, const bf16t* __restrict__ wvt,
    const float* __restrict__ bq, const float* __restrict__ bk, const float* __restrict__ bv,
    bf16t* __restrict__ qbf, bf16t* __restrict__ kbf, bf16t* __restrict__ vbf,
    bf16t* __restrict__ xn) {
  const int wave = threadIdx.x >> 5, lane = threadIdx.x & 31;
  const int ln = lane & 15, kh = (lane >> 4) & 1;
  const int which = blockIdx.y >> 2;                  // 0=q 1=k 2=v
  const int col0  = (blockIdx.y & 3) * 128;
  const int row0  = blockIdx.x * 128 + wave * 16;

  const bf16t* Bt   = (which == 0) ? wqt : (which == 1) ? wkt : wvt;
  const float* bias = (which == 0) ? bq  : (which == 1) ? bk  : bv;
  bf16t* outp       = (which == 0) ? qbf : (which == 1) ? kbf : vbf;

  const bf16t* arp = xbf + (size_t)(row0 + ln) * E_;

  v8f acc[8];
#pragma unroll
  for (int t = 0; t < 8; ++t)
#pragma unroll
    for (int i = 0; i < 8; ++i) acc[t][i] = 0.f;

  gemm_mainloop_ldsB(arp, Bt + (size_t)col0 * E_, E_, threadIdx.x, lane, acc);

#pragma unroll
  for (int t = 0; t < 8; ++t) {
    float bb = bias[col0 + t * 16 + ln];
#pragma unroll
    for (int i = 0; i < 8; ++i) acc[t][i] += bb;
  }

  // per-row 1/||.|| across this wave's 128 cols (one full head)
  float inv[8];
#pragma unroll
  for (int i = 0; i < 8; ++i) inv[i] = 1.f;
  if (which != 2) {
#pragma unroll
    for (int i = 0; i < 8; ++i) {
      float s = 0.f;
#pragma unroll
      for (int t = 0; t < 8; ++t) s += acc[t][i] * acc[t][i];
#pragma unroll
      for (int m = 1; m <= 8; m <<= 1) s += __shfl_xor(s, m, 32);
      inv[i] = 1.f / fmaxf(sqrtf(s), 1e-12f);
    }
  }

  const int h = col0 >> 7;
#pragma unroll
  for (int t = 0; t < 8; ++t) {
    int d = (col0 & 127) + t * 16 + ln;               // 0..127
#pragma unroll
    for (int i = 0; i < 8; ++i) {
      int r = row0 + i + 8 * kh;
      int b = r >> 12, c = r & (C_ - 1);
      float v = acc[t][i];
      outp[((size_t)(b * H_ + h) * C_ + c) * D_ + d] = (bf16t)v;
      if (which != 2) {
        size_t xi = ((size_t)(b * H_ + h) * TWOC + which * C_ + c) * (size_t)D_ + d;
        xn[xi] = (bf16t)(v * inv[i]);
      }
    }
  }
}

// ---------------------------------------------------------------------------
// 3) dists = xn @ means^T (per head) via WMMA; means staged in LDS (async);
//    store cluster-major dists for top-k; in-register argmax -> aux partial
//    (unit vectors: ||xn - routed||^2 = 2 - 2*best_dot).
// ---------------------------------------------------------------------------
__global__ __launch_bounds__(256) void dists_kernel(
    const bf16t* __restrict__ xn, const bf16t* __restrict__ meansbf,
    float* __restrict__ dstore, float* __restrict__ aux) {
  __shared__ bf16t smeans[NC_ * D_];                  // 16 KB, shared by 8 waves
  const int tid = threadIdx.x;
  const int wave = tid >> 5, lane = tid & 31;
  const int ln = lane & 15, kh = (lane >> 4) & 1;
  const int row0 = blockIdx.x * 128 + wave * 16;      // row in (B,H,2C)
  const int bh = row0 >> 13;                          // / 8192
  const int h = bh & (H_ - 1);
  const int l0 = row0 & (TWOC - 1);

  {
    const bf16t* g = meansbf + (size_t)h * NC_ * D_ + tid * 32;
#pragma unroll
    for (int j = 0; j < 4; ++j) async_copy16(g + j * 8, &smeans[tid * 32 + j * 8]);
  }
  async_wait_all();
  __syncthreads();

  const bf16t* arp = xn + (size_t)(row0 + ln) * D_;
  const bf16t* bp[4];
#pragma unroll
  for (int t = 0; t < 4; ++t) bp[t] = &smeans[(t * 16 + ln) * D_];

  v8f acc[4];
#pragma unroll
  for (int t = 0; t < 4; ++t)
#pragma unroll
    for (int i = 0; i < 8; ++i) acc[t][i] = 0.f;

#pragma unroll
  for (int k0 = 0; k0 < D_; k0 += 32) {
    v16bf a = a_frag(arp + k0, lane);
#pragma unroll
    for (int t = 0; t < 4; ++t)
      acc[t] = wmma_bf16(a, b_frag(bp[t] + k0, lane), acc[t]);
  }

  float bv[8];
#pragma unroll
  for (int i = 0; i < 8; ++i) bv[i] = -3.0e38f;

#pragma unroll
  for (int t = 0; t < 4; ++t) {
    int cl = t * 16 + ln;
#pragma unroll
    for (int i = 0; i < 8; ++i) {
      int l = l0 + i + 8 * kh;
      int qk = l >> 12, c = l & (C_ - 1);
      float v = acc[t][i];
      dstore[((size_t)(bh * 2 + qk) * NC_ + cl) * C_ + c] = v;
      bv[i] = fmaxf(bv[i], v);
    }
  }
#pragma unroll
  for (int i = 0; i < 8; ++i)
#pragma unroll
    for (int m = 1; m <= 8; m <<= 1) bv[i] = fmaxf(bv[i], __shfl_xor(bv[i], m, 32));

  if (ln == 0) {
    float part = 0.f;
#pragma unroll
    for (int i = 0; i < 8; ++i) part += 2.f - 2.f * bv[i];
    unsafeAtomicAdd(aux, part);
  }
}

// ---------------------------------------------------------------------------
// 4) top-128 of 4096 per (b,h,{q,k},cluster): LDS iterative argmax,
//    tie-break by smaller index, indices emitted in descending-value order.
// ---------------------------------------------------------------------------
__global__ __launch_bounds__(256) void topk_kernel(const float* __restrict__ dstore,
                                                   int* __restrict__ topk) {
  __shared__ float sv[C_];
  __shared__ float rv[256];
  __shared__ int   ri[256];
  const int tid = threadIdx.x;
  const size_t blk = blockIdx.x;                      // (bh*2+qk)*NC + cl
  const float* dv = dstore + blk * C_;
  int* outp = topk + blk * WSZ_;

  for (int j = tid; j < C_; j += 256) sv[j] = dv[j];
  __syncthreads();

  for (int it = 0; it < WSZ_; ++it) {
    float bv = -3.0e38f; int bi = 0x7fffffff;
    for (int j = tid; j < C_; j += 256) {
      float v = sv[j];
      if (v > bv || (v == bv && j < bi)) { bv = v; bi = j; }
    }
    rv[tid] = bv; ri[tid] = bi;
    __syncthreads();
    for (int s = 128; s > 0; s >>= 1) {
      if (tid < s) {
        float ov = rv[tid + s]; int oi = ri[tid + s];
        if (ov > rv[tid] || (ov == rv[tid] && oi < ri[tid])) { rv[tid] = ov; ri[tid] = oi; }
      }
      __syncthreads();
    }
    if (tid == 0) { outp[it] = ri[0]; sv[ri[0]] = -3.0e38f; }
    __syncthreads();
  }
}

// ---------------------------------------------------------------------------
// 5) windowed attention per (b,h,cluster): 128x128x128, all-WMMA.
//    QK^T fragments gathered from global via token indices (per-step B array
//    in distinct registers for pipelining); softmax in registers; exp staged
//    to LDS as the A operand of attn@V; V staged transposed in LDS. 64 KB LDS.
// ---------------------------------------------------------------------------
__global__ __launch_bounds__(256) void attn_kernel(
    const bf16t* __restrict__ qbf, const bf16t* __restrict__ kbf,
    const bf16t* __restrict__ vbf, const int* __restrict__ topk,
    float* __restrict__ numer, float* __restrict__ cnt) {
  __shared__ bf16t sc[WSZ_ * WSZ_];   // exp(scores) row-major (A operand)
  __shared__ bf16t vt[D_ * WSZ_];     // V transposed: vt[d][tok]
  const int tid = threadIdx.x, wave = tid >> 5, lane = tid & 31;
  const int ln = lane & 15, kh = (lane >> 4) & 1;
  const int blk = blockIdx.x;                         // bh*NC + cl
  const int bh = blk >> 6, cl = blk & (NC_ - 1);

  const int* qi = topk + ((size_t)(bh * 2 + 0) * NC_ + cl) * WSZ_;
  const int* ki = topk + ((size_t)(bh * 2 + 1) * NC_ + cl) * WSZ_;
  const bf16t* qb = qbf + (size_t)bh * C_ * D_;
  const bf16t* kb = kbf + (size_t)bh * C_ * D_;
  const bf16t* vb = vbf + (size_t)bh * C_ * D_;

  // stage V transposed into LDS: 2 threads per gathered row
  {
    int j = tid >> 1, d0 = (tid & 1) * 64;
    const bf16t* vrow = vb + (size_t)ki[j] * D_;
#pragma unroll
    for (int d = 0; d < 64; ++d) vt[(d0 + d) * WSZ_ + j] = vrow[d0 + d];
  }

  // ---- phase 1: scores = qs @ ks^T (gathered fragments from global) ----
  const int row0 = wave * 16;
  const bf16t* arow = qb + (size_t)qi[row0 + ln] * D_;
  const bf16t* krow[8];
#pragma unroll
  for (int t = 0; t < 8; ++t) krow[t] = kb + (size_t)ki[t * 16 + ln] * D_;

  v8f acc[8];
#pragma unroll
  for (int t = 0; t < 8; ++t)
#pragma unroll
    for (int i = 0; i < 8; ++i) acc[t][i] = 0.f;

#pragma unroll
  for (int k0 = 0; k0 < D_; k0 += 32) {
    v16bf a = a_frag(arow + k0, lane);
    v16bf bfr[8];
#pragma unroll
    for (int t = 0; t < 8; ++t) bfr[t] = b_frag(krow[t] + k0, lane);
#pragma unroll
    for (int t = 0; t < 8; ++t) acc[t] = wmma_bf16(a, bfr[t], acc[t]);
  }

  // ---- phase 2: softmax in registers (per-row over 128 cols) ----
  const float scl = 0.088388347648318447f;            // 1/sqrt(128)
  float mx[8], sum[8], rs[8];
#pragma unroll
  for (int i = 0; i < 8; ++i) {
    float m = -3.0e38f;
#pragma unroll
    for (int t = 0; t < 8; ++t) m = fmaxf(m, acc[t][i]);
#pragma unroll
    for (int s = 1; s <= 8; s <<= 1) m = fmaxf(m, __shfl_xor(m, s, 32));
    mx[i] = m; sum[i] = 0.f;
  }
#pragma unroll
  for (int t = 0; t < 8; ++t)
#pragma unroll
    for (int i = 0; i < 8; ++i) {
      float e = __expf((acc[t][i] - mx[i]) * scl);
      acc[t][i] = e;
      sum[i] += e;
    }
#pragma unroll
  for (int i = 0; i < 8; ++i) {
#pragma unroll
    for (int s = 1; s <= 8; s <<= 1) sum[i] += __shfl_xor(sum[i], s, 32);
    rs[i] = 1.f / sum[i];
  }

  // stage exp(scores) to LDS in row-major bf16 (A operand layout)
#pragma unroll
  for (int t = 0; t < 8; ++t)
#pragma unroll
    for (int i = 0; i < 8; ++i)
      sc[(row0 + i + 8 * kh) * WSZ_ + t * 16 + ln] = (bf16t)acc[t][i];
  __syncthreads();

  // ---- phase 3: bo = exp(scores) @ V (both operands in LDS) ----
  v8f acc2[8];
#pragma unroll
  for (int t = 0; t < 8; ++t)
#pragma unroll
    for (int i = 0; i < 8; ++i) acc2[t][i] = 0.f;

  const bf16t* arow2 = &sc[(row0 + ln) * WSZ_];
  const bf16t* vtp[8];
#pragma unroll
  for (int t = 0; t < 8; ++t) vtp[t] = &vt[(t * 16 + ln) * WSZ_];

#pragma unroll
  for (int k0 = 0; k0 < WSZ_; k0 += 32) {
    v16bf a = a_frag(arow2 + k0, lane);
    v16bf bfr[8];
#pragma unroll
    for (int t = 0; t < 8; ++t) bfr[t] = b_frag(vtp[t] + k0, lane);
#pragma unroll
    for (int t = 0; t < 8; ++t) acc2[t] = wmma_bf16(a, bfr[t], acc2[t]);
  }

  // ---- phase 4: scatter-add (tokens may repeat across clusters) ----
  int qt[8];
#pragma unroll
  for (int i = 0; i < 8; ++i) qt[i] = qi[row0 + i + 8 * kh];
#pragma unroll
  for (int t = 0; t < 8; ++t) {
    int d = t * 16 + ln;
#pragma unroll
    for (int i = 0; i < 8; ++i)
      unsafeAtomicAdd(numer + ((size_t)bh * C_ + qt[i]) * D_ + d, acc2[t][i] * rs[i]);
  }
  if (tid < WSZ_) unsafeAtomicAdd(cnt + (size_t)bh * C_ + qi[tid], 1.0f);
}

// ---------------------------------------------------------------------------
// 6) A = (numer/denom) transposed to (B,C,H*D), cast bf16
// ---------------------------------------------------------------------------
__global__ void finalize_a_kernel(const float* __restrict__ numer,
                                  const float* __restrict__ cnt,
                                  bf16t* __restrict__ abf) {
  size_t idx = (size_t)blockIdx.x * 256 + threadIdx.x;  // B*C*HD
  int n = (int)(idx & 511);
  size_t bc = idx >> 9;
  int c = (int)(bc & (C_ - 1)), b = (int)(bc >> 12);
  int h = n >> 7, d = n & 127;
  size_t bhc = (size_t)(b * H_ + h) * C_ + c;
  float den = cnt[bhc] + 1e-5f;
  abf[idx] = (bf16t)(numer[bhc * D_ + d] / den);
}

// ---------------------------------------------------------------------------
// 7) output GEMM: [32768 x 512] x Wu[512 x 512] + bu -> d_out (f32)
// ---------------------------------------------------------------------------
__global__ __launch_bounds__(256) void gemm_out_kernel(
    const bf16t* __restrict__ abf, const bf16t* __restrict__ wut,
    const float* __restrict__ bu, float* __restrict__ out) {
  const int wave = threadIdx.x >> 5, lane = threadIdx.x & 31;
  const int ln = lane & 15, kh = (lane >> 4) & 1;
  const int col0 = blockIdx.y * 128;
  const int row0 = blockIdx.x * 128 + wave * 16;

  const bf16t* arp = abf + (size_t)(row0 + ln) * HD_;

  v8f acc[8];
#pragma unroll
  for (int t = 0; t < 8; ++t)
#pragma unroll
    for (int i = 0; i < 8; ++i) acc[t][i] = 0.f;

  gemm_mainloop_ldsB(arp, wut + (size_t)col0 * HD_, HD_, threadIdx.x, lane, acc);

#pragma unroll
  for (int t = 0; t < 8; ++t) {
    int n = col0 + t * 16 + ln;
    float bb = bu[n];
#pragma unroll
    for (int i = 0; i < 8; ++i) {
      int r = row0 + i + 8 * kh;
      out[(size_t)r * E_ + n] = acc[t][i] + bb;
    }
  }
}

__global__ void write_aux_kernel(const float* aux, float* out_aux) {
  // aux_loss = mean((xn - routed)^2) * COMMITMENT ; mean over B*H*2C*D = 33554432
  out_aux[0] = aux[0] * (1.0e-4f / 33554432.0f);
}

// ---------------------------------------------------------------------------
// host launcher
// ---------------------------------------------------------------------------
extern "C" void kernel_launch(void* const* d_in, const int* in_sizes, int n_in,
                              void* d_out, int out_size, void* d_ws, size_t ws_size,
                              hipStream_t stream) {
  (void)in_sizes; (void)n_in; (void)out_size; (void)ws_size;
  const float* x     = (const float*)d_in[0];
  // d_in[1] = attention_mask (all-ones; unused by the reference math)
  const float* Wq    = (const float*)d_in[2];
  const float* bq    = (const float*)d_in[3];
  const float* Wk    = (const float*)d_in[4];
  const float* bk    = (const float*)d_in[5];
  const float* Wv    = (const float*)d_in[6];
  const float* bv    = (const float*)d_in[7];
  const float* Wu    = (const float*)d_in[8];
  const float* bu    = (const float*)d_in[9];
  const float* means = (const float*)d_in[10];
  float* out = (float*)d_out;

  char* ws = (char*)d_ws;
  size_t off = 0;
  auto take = [&](size_t bytes) -> char* {
    char* p = ws + off;
    off = (off + bytes + 255) & ~(size_t)255;
    return p;
  };

  bf16t* xbf     = (bf16t*)take((size_t)MTOK * E_ * sizeof(bf16t));
  bf16t* wqt     = (bf16t*)take((size_t)E_ * HD_ * sizeof(bf16t));
  bf16t* wkt     = (bf16t*)take((size_t)E_ * HD_ * sizeof(bf16t));
  bf16t* wvt     = (bf16t*)take((size_t)E_ * HD_ * sizeof(bf16t));
  bf16t* wut     = (bf16t*)take((size_t)HD_ * E_ * sizeof(bf16t));
  bf16t* meansbf = (bf16t*)take((size_t)H_ * NC_ * D_ * sizeof(bf16t));
  bf16t* qbf     = (bf16t*)take((size_t)B_ * H_ * C_ * D_ * sizeof(bf16t));
  bf16t* kbf     = (bf16t*)take((size_t)B_ * H_ * C_ * D_ * sizeof(bf16t));
  bf16t* vbf     = (bf16t*)take((size_t)B_ * H_ * C_ * D_ * sizeof(bf16t));
  bf16t* xn      = (bf16t*)take((size_t)B_ * H_ * TWOC * D_ * sizeof(bf16t));
  float* dstore  = (float*)take((size_t)B_ * H_ * 2 * NC_ * C_ * sizeof(float));
  int*   topk    = (int*)  take((size_t)B_ * H_ * 2 * NC_ * WSZ_ * sizeof(int));
  float* numer   = (float*)take((size_t)B_ * H_ * C_ * D_ * sizeof(float));
  float* cnt     = (float*)take((size_t)B_ * H_ * C_ * sizeof(float));
  float* aux     = (float*)take(256);
  bf16t* abf     = (bf16t*)take((size_t)MTOK * HD_ * sizeof(bf16t));

  // 0) zero accumulators
  init_zero_kernel<<<65536, 256, 0, stream>>>(numer, cnt, aux);
  // 1) casts
  cast_weights_kernel<<<4224, 256, 0, stream>>>(Wq, Wk, Wv, Wu, means,
                                                wqt, wkt, wvt, wut, meansbf);
  cast_x_kernel<<<65536, 256, 0, stream>>>(x, xbf);
  // 2) QKV projections (grid.y: 3 matrices x 4 col-tiles)
  gemm_qkv_kernel<<<dim3(MTOK / 128, 12), 256, 0, stream>>>(
      xbf, wqt, wkt, wvt, bq, bk, bv, qbf, kbf, vbf, xn);
  // 3) routing dists + aux loss
  dists_kernel<<<(B_ * H_ * TWOC) / 128, 256, 0, stream>>>(xn, meansbf, dstore, aux);
  // 4) top-k token selection per cluster (q and k)
  topk_kernel<<<B_ * H_ * 2 * NC_, 256, 0, stream>>>(dstore, topk);
  // 5) clustered attention + scatter
  attn_kernel<<<B_ * H_ * NC_, 256, 0, stream>>>(qbf, kbf, vbf, topk, numer, cnt);
  // 6) average + transpose + cast
  finalize_a_kernel<<<65536, 256, 0, stream>>>(numer, cnt, abf);
  // 7) output projection
  gemm_out_kernel<<<dim3(MTOK / 128, 4), 256, 0, stream>>>(abf, wut, bu, out);
  // 8) aux scalar
  write_aux_kernel<<<1, 1, 0, stream>>>(aux, out + (size_t)B_ * C_ * E_);
}
